// SeqCondAttention_54468775248201
// MI455X (gfx1250) — matrix operand order repaired and checked by reading, not compiled, not run on hardware
//
#include <hip/hip_runtime.h>
#include <hip/hip_bf16.h>
#include <math.h>

// ---------------- static problem config (matches reference) ----------------
#define BB 4
#define LL 4096
#define DD 1024
#define KK 32
#define HH 32
#define AA 4
#define CKW 4
#define DI 1024           // d_inner
#define ZC 2080           // 2*DI + KK
#define BL (BB * LL)      // 16384 rows
#define NCHUNK 16
#define CHUNK (LL / NCHUNK)   // 256

typedef __attribute__((ext_vector_type(16))) __bf16 v16bf;
typedef __attribute__((ext_vector_type(8)))  __bf16 bf16x8;
typedef __attribute__((ext_vector_type(8)))  float  v8f;

// ---------------- workspace layout (bytes) ----------------
static constexpr size_t OFF_XB    = 0;                                        // x bf16 [BL, DD]
static constexpr size_t OFF_WINT  = OFF_XB    + (size_t)BL * DD * 2;          // W_in^T bf16 [ZC, DD]
static constexpr size_t OFF_WOUTT = OFF_WINT  + (size_t)ZC * DD * 2;          // W_out^T bf16 [DD, DD]
static constexpr size_t OFF_Z     = OFF_WOUTT + (size_t)DD * DD * 2;          // z f32 [BL, ZC]
static constexpr size_t OFF_MERG  = OFF_Z     + (size_t)BL * ZC * 4;          // merged/cs f32 [BL, ZC]
static constexpr size_t OFF_GATE  = OFF_MERG  + (size_t)BL * ZC * 4;          // gate f32 [BL, DI]
static constexpr size_t OFF_CSUM  = OFF_GATE  + (size_t)BL * DI * 4;          // chunk sums f32 [B*ZC, NCHUNK]
static constexpr size_t OFF_YG    = OFF_CSUM  + (size_t)BB * ZC * NCHUNK * 4; // y_gated bf16 [BL, DI]

// ---------------- elementwise converts ----------------
__global__ __launch_bounds__(256) void cvt_bf16(const float* __restrict__ src,
                                                __bf16* __restrict__ dst, int n) {
  int i = blockIdx.x * 256 + threadIdx.x;
  if (i < n) dst[i] = (__bf16)src[i];
}

// src is [Kd, N] row-major f32; dst is [N, Kd] row-major bf16 (so GEMM B-columns
// become contiguous, enabling b128 fragment loads).
__global__ __launch_bounds__(256) void transpose_cvt_bf16(const float* __restrict__ src,
                                                          __bf16* __restrict__ dst,
                                                          int Kd, int N) {
  int i = blockIdx.x * 256 + threadIdx.x;
  if (i >= N * Kd) return;
  int n = i / Kd, kd = i - n * Kd;
  dst[i] = (__bf16)src[(size_t)kd * N + n];
}

// ---------------- WMMA bf16 GEMM: C[M,N] f32 = A[M,Kd] * B[Kd,N] ----------------
// Bt is B transposed ([N,Kd] bf16). One wave computes a 32x32 macro-tile
// (2 A-frags x 2 B-frags -> 4 v_wmma per k-step; halves fragment traffic per
// WMMA and gives 4 independent accumulation chains to hide XDL latency).
// Fragment layouts per CDNA5 ISA 7.12.2 (16-bit A 16x32; B column-per-lane).
__global__ __launch_bounds__(128)
void gemm_bf16_wmma(const __bf16* __restrict__ A, const __bf16* __restrict__ Bt,
                    float* __restrict__ C, int Ncols, int Kd) {
  const int lane = threadIdx.x;                  // 0..31
  const int wave = threadIdx.y;                  // 0..3
  const int ntn  = Ncols >> 5;                   // 32-wide macro-tiles
  const int tile = blockIdx.x * 4 + wave;
  const int tm   = (tile / ntn) << 5;
  const int tn   = (tile % ntn) << 5;
  const int hi   = lane >> 4;                    // lane half
  const int lo   = lane & 15;

  // A: lane holds row tm+lo (and tm+16+lo); K pairs {hi*8..+7} and {16+hi*8..}
  const __bf16* aRow0 = A + (size_t)(tm + lo) * Kd + hi * 8;
  const __bf16* aRow1 = aRow0 + (size_t)16 * Kd;
  // B: lane holds column tn+lo (and tn+16+lo); K {hi*8..+7} and {16+hi*8..}
  const __bf16* bRow0 = Bt + (size_t)(tn + lo) * Kd + hi * 8;
  const __bf16* bRow1 = bRow0 + (size_t)16 * Kd;

  union Frag { v16bf v; bf16x8 h[2]; };
  Frag a0, a1, b0, b1;
  v8f c00 = {}, c01 = {}, c10 = {}, c11 = {};
  for (int kk = 0; kk < Kd; kk += 32) {
    __builtin_prefetch(aRow0 + kk + 64, 0, 1);   // global_prefetch_b8
    __builtin_prefetch(aRow1 + kk + 64, 0, 1);
    __builtin_prefetch(bRow0 + kk + 64, 0, 1);
    __builtin_prefetch(bRow1 + kk + 64, 0, 1);
    a0.h[0] = *(const bf16x8*)(aRow0 + kk);
    a0.h[1] = *(const bf16x8*)(aRow0 + kk + 16);
    a1.h[0] = *(const bf16x8*)(aRow1 + kk);
    a1.h[1] = *(const bf16x8*)(aRow1 + kk + 16);
    b0.h[0] = *(const bf16x8*)(bRow0 + kk);
    b0.h[1] = *(const bf16x8*)(bRow0 + kk + 16);
    b1.h[0] = *(const bf16x8*)(bRow1 + kk);
    b1.h[1] = *(const bf16x8*)(bRow1 + kk + 16);
    c00 = __builtin_amdgcn_wmma_f32_16x16x32_bf16(false, a0.v, false, b0.v, (short)0, c00, false, false);
    c01 = __builtin_amdgcn_wmma_f32_16x16x32_bf16(false, a0.v, false, b1.v, (short)0, c01, false, false);
    c10 = __builtin_amdgcn_wmma_f32_16x16x32_bf16(false, a1.v, false, b0.v, (short)0, c10, false, false);
    c11 = __builtin_amdgcn_wmma_f32_16x16x32_bf16(false, a1.v, false, b1.v, (short)0, c11, false, false);
  }
  // C/D layout: VGPR r -> row r + hi*8, column lo
  float* cr0 = C + (size_t)(tm + hi * 8) * Ncols + (tn + lo);
  float* cr1 = C + (size_t)(tm + 16 + hi * 8) * Ncols + (tn + lo);
#pragma unroll
  for (int r = 0; r < 8; ++r) {
    cr0[(size_t)r * Ncols]      = c00[r];
    cr0[(size_t)r * Ncols + 16] = c01[r];
    cr1[(size_t)r * Ncols]      = c10[r];
    cr1[(size_t)r * Ncols + 16] = c11[r];
  }
}

// ---------------- causal depthwise conv + gate + phase features ----------------
// The 4 causal z-rows are staged into LDS via CDNA5 async global->LDS copies
// (GLOBAL_LOAD_ASYNC_TO_LDS_B128, ASYNCcnt) bypassing the VGPR path entirely.
__global__ __launch_bounds__(256)
void conv_features(const float* __restrict__ z, const float* __restrict__ ck,
                   const float* __restrict__ theta, const float* __restrict__ dslope,
                   const float* __restrict__ aslope, const float* __restrict__ sscale,
                   float* __restrict__ merged, float* __restrict__ gate) {
  __shared__ float zst[CKW * ZC];    // staged raw z rows (33,280 B)
  __shared__ float zc[ZC];           // conv result
  __shared__ float pw[KK];
  const int bl = blockIdx.x;
  const int b  = bl >> 12;           // / LL
  const int l  = bl & (LL - 1);
  const int t  = threadIdx.x;

  // --- async-stage the 4 causal rows (zero-fill before sequence start) ---
#pragma unroll
  for (int j = 0; j < CKW; ++j) {
    const int ll = l - (CKW - 1) + j;
    if (ll >= 0) {
      const float* src = z + ((size_t)(b * LL + ll)) * ZC;
      for (int c4 = t; c4 < ZC / 4; c4 += 256) {        // 520 x b128 per row
        uint64_t ga   = (uint64_t)(uintptr_t)(src + c4 * 4);
        uint32_t loff = (uint32_t)(uintptr_t)(&zst[j * ZC + c4 * 4]);
        asm volatile("global_load_async_to_lds_b128 %0, %1, off"
                     :: "v"(loff), "v"(ga) : "memory");
      }
    } else {
      for (int c = t; c < ZC; c += 256) zst[j * ZC + c] = 0.f;
    }
  }
  asm volatile("s_wait_asynccnt 0" ::: "memory");
  __syncthreads();

  // --- depthwise causal conv from LDS ---
  for (int c = t; c < ZC; c += 256) {
    float acc = 0.f;
#pragma unroll
    for (int j = 0; j < CKW; ++j) acc += zst[j * ZC + c] * ck[j * ZC + c];
    zc[c] = acc;
  }
  __syncthreads();

  const size_t base = (size_t)bl * ZC;
  if (t < KK) {
    float s_raw = zc[2 * DI + t];
    float sl = (t < KK - AA) ? dslope[t] : aslope[t - (KK - AA)];
    float sp = log1pf(expf(sl));                       // softplus
    float dist = (t < KK - AA) ? (float)(LL - 1 - l) : (float)l;
    float p = expf(sscale[t] * s_raw - sp * dist);
    pw[t] = p;
    merged[base + t] = p;
  }
  __syncthreads();

  for (int c = t; c < DI; c += 256) {
    int k = c >> 5;
    float xv  = zc[c];
    float phi = xv * theta[c];
    float s, co;
    __sincosf(phi, &s, &co);
    float p = pw[k];
    merged[base + KK + c]      = p * co;
    merged[base + KK + DI + c] = p * s;
    float g = zc[DI + c];
    gate[(size_t)bl * DI + c] = g / (1.f + expf(-g));  // silu
  }
}

// ---------------- chunked prefix scan over L (3-phase, phase 3 fused below) ----
__global__ __launch_bounds__(256)
void scan_chunks_local(float* __restrict__ merged, float* __restrict__ csum) {
  int t = blockIdx.x * 256 + threadIdx.x;   // t < BB*NCHUNK*ZC (exact)
  int c = t % ZC;
  int r = t / ZC;
  int chunk = r % NCHUNK;
  int b = r / NCHUNK;
  float run = 0.f;
  size_t idx = ((size_t)(b * LL + chunk * CHUNK)) * ZC + c;
  for (int i = 0; i < CHUNK; ++i) {
    run += merged[idx];
    merged[idx] = run;
    idx += ZC;
  }
  csum[((size_t)(b * ZC + c)) * NCHUNK + chunk] = run;
}

__global__ __launch_bounds__(256)
void scan_chunk_offsets(float* __restrict__ csum) {
  int t = blockIdx.x * 256 + threadIdx.x;
  if (t >= BB * ZC) return;
  size_t base = (size_t)t * NCHUNK;
  float run = 0.f;
  for (int ch = 0; ch < NCHUNK; ++ch) {
    float v = csum[base + ch];
    csum[base + ch] = run;   // exclusive offset
    run += v;
  }
}

// ---------------- normalize + 32x32 head matmuls + gate -> bf16 ----------------
__global__ __launch_bounds__(256)
void finalize(const float* __restrict__ merged, const float* __restrict__ csum,
              const float* __restrict__ Wre, const float* __restrict__ Wim,
              const float* __restrict__ nscale, const float* __restrict__ gate,
              __bf16* __restrict__ ygated) {
  __shared__ float invden[KK], rsh[KK], msq[KK];
  __shared__ float reL[DI], imL[DI];
  __shared__ float wre[KK * HH], wim[KK * HH];
  __shared__ float nsr[HH], nsi[HH];
  const int bl = blockIdx.x;
  const int b  = bl >> 12;
  const int l  = bl & (LL - 1);
  const int t  = threadIdx.x;
  const int chunk = l >> 8;                    // l / CHUNK
  const size_t base  = (size_t)bl * ZC;
  const size_t cbase = (size_t)b * ZC;

  for (int i = t; i < KK * HH; i += 256) { wre[i] = Wre[i]; wim[i] = Wim[i]; }
  if (t < HH) { nsr[t] = nscale[t]; nsi[t] = nscale[HH + t]; }
  if (t < KK) {
    msq[t] = 0.f;
    float den = merged[base + t] + csum[(cbase + t) * NCHUNK + chunk];
    invden[t] = 1.f / fmaxf(den, 1e-4f);
  }
  __syncthreads();

  for (int c = t; c < DI; c += 256) {
    int k = c >> 5;
    float re = (merged[base + KK + c]      + csum[(cbase + KK + c)      * NCHUNK + chunk]) * invden[k];
    float im = (merged[base + KK + DI + c] + csum[(cbase + KK + DI + c) * NCHUNK + chunk]) * invden[k];
    reL[c] = re;
    imL[c] = im;
    atomicAdd(&msq[k], re * re + im * im);     // ds_add_f32
  }
  __syncthreads();
  if (t < KK) rsh[t] = rsqrtf(msq[t] / (2.f * HH) + 1e-5f);
  __syncthreads();

  const size_t gbase = (size_t)bl * DI;
  for (int c = t; c < DI; c += 256) {
    int k = c >> 5, h = c & 31;
    const float* rk = &reL[k << 5];
    const float* ik = &imL[k << 5];
    float acc = 0.f;
#pragma unroll 8
    for (int s = 0; s < HH; ++s)
      acc += rk[s] * nsr[s] * wre[(s << 5) + h] + ik[s] * nsi[s] * wim[(s << 5) + h];
    float y = acc * rsh[k];
    ygated[gbase + c] = (__bf16)(y * gate[gbase + c]);
  }
}

// ---------------- host-side launch ----------------
extern "C" void kernel_launch(void* const* d_in, const int* in_sizes, int n_in,
                              void* d_out, int out_size, void* d_ws, size_t ws_size,
                              hipStream_t stream) {
  const float* x      = (const float*)d_in[0];
  const float* W_in   = (const float*)d_in[1];
  const float* convk  = (const float*)d_in[2];
  const float* theta  = (const float*)d_in[3];
  const float* dsl    = (const float*)d_in[4];
  const float* asl    = (const float*)d_in[5];
  const float* ssc    = (const float*)d_in[6];
  const float* Wre    = (const float*)d_in[7];
  const float* Wim    = (const float*)d_in[8];
  const float* nsc    = (const float*)d_in[9];
  const float* W_out  = (const float*)d_in[10];
  float* out = (float*)d_out;

  char* ws = (char*)d_ws;
  __bf16* xb    = (__bf16*)(ws + OFF_XB);
  __bf16* winT  = (__bf16*)(ws + OFF_WINT);
  __bf16* woutT = (__bf16*)(ws + OFF_WOUTT);
  float*  zbuf  = (float*) (ws + OFF_Z);
  float*  merg  = (float*) (ws + OFF_MERG);
  float*  gate  = (float*) (ws + OFF_GATE);
  float*  csum  = (float*) (ws + OFF_CSUM);
  __bf16* yg    = (__bf16*)(ws + OFF_YG);

  // 1) precision conversion + weight transposes for the WMMA GEMMs
  cvt_bf16<<<(BL * DD) / 256, 256, 0, stream>>>(x, xb, BL * DD);
  transpose_cvt_bf16<<<(ZC * DD) / 256, 256, 0, stream>>>(W_in, winT, DD, ZC);
  transpose_cvt_bf16<<<(DD * DD) / 256, 256, 0, stream>>>(W_out, woutT, DD, DD);

  // 2) z = x @ W_in   (16384x2080, K=1024) — tensor path, 32x32 per wave
  {
    int tiles = (BL / 32) * (ZC / 32);   // 512*65 = 33280, divisible by 4
    gemm_bf16_wmma<<<tiles / 4, dim3(32, 4), 0, stream>>>(xb, winT, zbuf, ZC, DD);
  }

  // 3) depthwise conv (async-LDS staged) + gate + score weights + phase features
  conv_features<<<BL, 256, 0, stream>>>(zbuf, convk, theta, dsl, asl, ssc, merg, gate);

  // 4) causal cumsum over L (chunked parallel scan; offsets folded into finalize)
  scan_chunks_local<<<(BB * NCHUNK * ZC) / 256, 256, 0, stream>>>(merg, csum);
  scan_chunk_offsets<<<(BB * ZC + 255) / 256, 256, 0, stream>>>(csum);

  // 5) normalize + per-head 32x32 matmuls + gate -> bf16 activations
  finalize<<<BL, 256, 0, stream>>>(merg, csum, Wre, Wim, nsc, gate, yg);

  // 6) out = y_gated @ W_out  (16384x1024, K=1024) — tensor path, f32 out
  {
    int tiles = (BL / 32) * (DD / 32);   // 512*32 = 16384
    gemm_bf16_wmma<<<tiles / 4, dim3(32, 4), 0, stream>>>(yg, woutT, out, DD, DD);
  }
}
